// SuperpixelConv_48584670052562
// MI455X (gfx1250) — matrix operand order repaired and checked by reading, not compiled, not run on hardware
//
#include <hip/hip_runtime.h>

// ---------------------------------------------------------------------------
// SuperpixelConv on gfx1250: split-precision V_WMMA_F32_16X16X32_BF16 with
// weight slabs staged into LDS by the Tensor Data Mover (double buffered).
//   out[b,o,p] = sum_k sum_i r[b,k,p] * x[b,i,p+off(k)] * Wk[o,i,k] + bias[o]
// 49 shifted 64x64 pointwise GEMMs with r folded into x; f32 operands split
// hi+lo bf16 (bf16x3): D = Whi*Xhi + Whi*Xlo + Wlo*Xhi, f32 accumulate.
// Per-k 16KB weight slab is shared by all 8 waves of a block via LDS (TDM),
// cutting L2 weight re-fetch 8x; A-operand fetch is ds_load_b128.
// ---------------------------------------------------------------------------

typedef __bf16 v16bf __attribute__((ext_vector_type(16)));
typedef float  v8f   __attribute__((ext_vector_type(8)));
typedef unsigned int v4u __attribute__((ext_vector_type(4)));
typedef int    v8s   __attribute__((ext_vector_type(8)));
typedef int    v4s   __attribute__((ext_vector_type(4)));

#define BSZ   2
#define CCH   64
#define HH    96
#define WW    96
#define KK    7
#define K2    49
#define HW    (HH * WW)          // 9216
#define FANIN (CCH * K2)         // 3136
#define FS    9

#define R_ELEMS   (BSZ * K2 * HW)    // 903168 f32 for r (~3.6 MB)
#define SLAB_BF16 8192               // per-k slab: 4096 hi + 4096 lo bf16
#define SLAB_BYTES 16384
#define WT2_ELEMS (K2 * SLAB_BF16)   // 401408 bf16 (~0.8 MB)
#define WTILES    (WW / 16)          // 6
#define NTILES    (BSZ * HH * WTILES) // 1152 pixel tiles

// ---- pack W[o, i*49+k] into per-k contiguous slabs, WMMA A-operand order --
// Slab k (16KB): [hi: 8 tiles x 512 bf16][lo: 8 tiles x 512 bf16]
// tile tt = cc*4 + nt (cc = 32-chan chunk, nt = M tile); within a tile lane
// l's 16 bf16 elements are contiguous (32B):
//   element e (vgpr v = e/2, halfword hw = e&1):
//   M = nt*16 + l%16 ; K = (v>=4?16:0) + (l/16)*8 + (v&3)*2 + hw
//   i = cc*32 + K    (16-bit A 16x32 layout)
__global__ void pack_w_kernel(const float* __restrict__ W,
                              __bf16* __restrict__ Wt2) {
    int t = blockIdx.x * 256 + threadIdx.x;   // one per (k,tile,lane,e)
    if (t >= K2 * 4096) return;               // 200704
    int e  = t & 15;
    int l  = (t >> 4) & 31;
    int tt = (t >> 9) & 7;
    int k  = t >> 12;                         // 0..48
    int nt = tt & 3;
    int cc = tt >> 2;

    int v  = e >> 1;
    int hw = e & 1;
    int K  = ((v >= 4) ? 16 : 0) + ((l >> 4) << 3) + ((v & 3) << 1) + hw;
    int o  = nt * 16 + (l & 15);
    int i  = cc * 32 + K;

    float  w  = W[o * FANIN + i * K2 + k];
    __bf16 hi = (__bf16)w;
    __bf16 lo = (__bf16)(w - (float)hi);
    int dst = k * SLAB_BF16 + tt * 512 + l * 16 + e;
    Wt2[dst]        = hi;
    Wt2[dst + 4096] = lo;
}

// ---- r[b,k,h,w] = sum_f sims[b,f,h,w] * sims_padded[b,f,h+dh,w+dw] --------
__global__ void compute_r_kernel(const float* __restrict__ sims,
                                 float* __restrict__ r) {
    int t = blockIdx.x * 256 + threadIdx.x;
    if (t >= BSZ * HW) return;
    int b = t / HW;
    int p = t - b * HW;
    int h = p / WW;
    int w = p - h * WW;
    const float* sb = sims + b * FS * HW;
    float s[FS];
#pragma unroll
    for (int f = 0; f < FS; ++f) s[f] = sb[f * HW + p];
    for (int k = 0; k < K2; ++k) {
        int dh = k / KK - 3;
        int dw = k % KK - 3;
        int hh = h + dh, ww = w + dw;
        float acc = 0.f;
        if (hh >= 0 && hh < HH && ww >= 0 && ww < WW) {
            int q = hh * WW + ww;
#pragma unroll
            for (int f = 0; f < FS; ++f) acc += s[f] * sb[f * HW + q];
        }
        r[(b * K2 + k) * HW + p] = acc;
    }
}

// ---- TDM: issue a 16KB 1-D tensor_load_to_lds of slab k ------------------
__device__ __forceinline__ void tdm_load_slab(unsigned long long gaddr,
                                              unsigned lds_off) {
    v4u g0;
    g0.x = 1u;                                      // count = 1, user desc
    g0.y = lds_off;                                 // lds_addr (bytes)
    g0.z = (unsigned)(gaddr & 0xffffffffu);         // global_addr[31:0]
    g0.w = (unsigned)((gaddr >> 32) & 0x1ffffffu)   // global_addr[56:32]
         | (2u << 30);                              // type = 2 (image)
    v8s g1;
    g1[0] = (int)(2u << 16);        // workgroup_mask=0, data_size=2 (4B)
    g1[1] = (int)(4096u << 16);     // tensor_dim0 = 4096 dwords (low 16)
    g1[2] = (int)(1u << 16);        // tensor_dim0 hi = 0; tensor_dim1 = 1
    g1[3] = (int)(4096u << 16);     // tensor_dim1 hi = 0; tile_dim0 = 4096
    g1[4] = 1;                      // tile_dim1 = 1, tile_dim2 = 0
    g1[5] = 4096;                   // tensor_dim0_stride = 4096 (low 32)
    g1[6] = 0;                      // stride0 hi, stride1 lo
    g1[7] = 0;
    v4s g2 = (v4s)0;
    v4s g3 = (v4s)0;
    v8s g4 = (v8s)0;                // unused extra group (non-cluster, 1-D)
    __builtin_amdgcn_tensor_load_to_lds(g0, g1, g2, g3, g4, 0);
}

// ---- main kernel: one wave = one 16-pixel row tile x 64 out channels ------
__global__ void __launch_bounds__(256, 1)
sconv_wmma_kernel(const float* __restrict__ x,
                  const float* __restrict__ r,
                  const __bf16* __restrict__ Wt2,
                  const float* __restrict__ bias,
                  float* __restrict__ out) {
    __shared__ __align__(128) unsigned char lds_w[2 * SLAB_BYTES];

    const int lane = threadIdx.x & 31;
    const int wave = threadIdx.x >> 5;
    const int lm = lane & 15;   // pixel column (B/C column N) / A row M
    const int lh = lane >> 4;   // lane half -> K group

    int tile = blockIdx.x * 8 + wave;   // 0..1151, grid sized exactly
    int b  = tile / (HH * WTILES);
    int rm = tile - b * (HH * WTILES);
    int h  = rm / WTILES;
    int w0 = (rm - h * WTILES) << 4;

    const float* xb = x + b * CCH * HW;
    const float* rb = r + b * K2 * HW;

    const unsigned lds_off = (unsigned)(unsigned long long)(const void*)lds_w;
    const unsigned long long wbase = (unsigned long long)(const void*)Wt2;

    v8f acc0 = {};
    v8f acc1 = {};
    v8f acc2 = {};
    v8f acc3 = {};

    // Prologue: stage slab 0.
    if (wave == 0) tdm_load_slab(wbase, lds_off);

    for (int k = 0; k < K2; ++k) {
        if (wave == 0) __builtin_amdgcn_s_wait_tensorcnt(0);  // slab k arrived
        __syncthreads();   // broadcast arrival; also WAR guard for (k+1)&1 buffer
        if (wave == 0 && k + 1 < K2)
            tdm_load_slab(wbase + (unsigned long long)(k + 1) * SLAB_BYTES,
                          lds_off + ((k + 1) & 1) * SLAB_BYTES);

        int dh = k / KK - 3;
        int dw = k % KK - 3;
        int hh = h + dh;
        int ww = w0 + lm + dw;                 // this lane's shifted pixel col
        bool inb = (hh >= 0) & (hh < HH) & (ww >= 0) & (ww < WW);
        float rv = rb[k * HW + h * WW + w0 + lm];   // per-pixel reweight
        const float* xp = xb + hh * WW + ww;        // + channel*HW per fetch

        // A-operand source: LDS (keeps addrspace(3) -> ds_load_b128 pairs).
        const v16bf* shi = (const v16bf*)(lds_w + (k & 1) * SLAB_BYTES);
        const v16bf* slo = shi + 256;               // +8192 bytes: lo plane

#pragma unroll
        for (int cc = 0; cc < 2; ++cc) {
            // B operand (32x16 bf16): lane l element e -> K = lh*8+e+(e>=8?8:0), N = lm
            int c0 = cc * 32 + lh * 8;
            v16bf Bhi, Blo;
#pragma unroll
            for (int e = 0; e < 16; ++e) {
                float xv = inb ? xp[(c0 + e + ((e >= 8) ? 8 : 0)) * HW] : 0.f;
                float bv = xv * rv;
                __bf16 bh = (__bf16)bv;
                Bhi[e] = bh;
                Blo[e] = (__bf16)(bv - (float)bh);
            }

            int tb = cc * 4;   // slab tile base for nt=0
            {
                v16bf Ahi = shi[(tb + 0) * 32 + lane];
                v16bf Alo = slo[(tb + 0) * 32 + lane];
                acc0 = __builtin_amdgcn_wmma_f32_16x16x32_bf16(
                    false, Ahi, false, Bhi, (short)0, acc0, false, false);
                acc0 = __builtin_amdgcn_wmma_f32_16x16x32_bf16(
                    false, Ahi, false, Blo, (short)0, acc0, false, false);
                acc0 = __builtin_amdgcn_wmma_f32_16x16x32_bf16(
                    false, Alo, false, Bhi, (short)0, acc0, false, false);
            }
            {
                v16bf Ahi = shi[(tb + 1) * 32 + lane];
                v16bf Alo = slo[(tb + 1) * 32 + lane];
                acc1 = __builtin_amdgcn_wmma_f32_16x16x32_bf16(
                    false, Ahi, false, Bhi, (short)0, acc1, false, false);
                acc1 = __builtin_amdgcn_wmma_f32_16x16x32_bf16(
                    false, Ahi, false, Blo, (short)0, acc1, false, false);
                acc1 = __builtin_amdgcn_wmma_f32_16x16x32_bf16(
                    false, Alo, false, Bhi, (short)0, acc1, false, false);
            }
            {
                v16bf Ahi = shi[(tb + 2) * 32 + lane];
                v16bf Alo = slo[(tb + 2) * 32 + lane];
                acc2 = __builtin_amdgcn_wmma_f32_16x16x32_bf16(
                    false, Ahi, false, Bhi, (short)0, acc2, false, false);
                acc2 = __builtin_amdgcn_wmma_f32_16x16x32_bf16(
                    false, Ahi, false, Blo, (short)0, acc2, false, false);
                acc2 = __builtin_amdgcn_wmma_f32_16x16x32_bf16(
                    false, Alo, false, Bhi, (short)0, acc2, false, false);
            }
            {
                v16bf Ahi = shi[(tb + 3) * 32 + lane];
                v16bf Alo = slo[(tb + 3) * 32 + lane];
                acc3 = __builtin_amdgcn_wmma_f32_16x16x32_bf16(
                    false, Ahi, false, Bhi, (short)0, acc3, false, false);
                acc3 = __builtin_amdgcn_wmma_f32_16x16x32_bf16(
                    false, Ahi, false, Blo, (short)0, acc3, false, false);
                acc3 = __builtin_amdgcn_wmma_f32_16x16x32_bf16(
                    false, Alo, false, Bhi, (short)0, acc3, false, false);
            }
        }
    }

    // Epilogue: D lane l, vgpr v -> channel m = v + 8*lh (within nt tile),
    // pixel n = lm. Stores are 64B-contiguous across lanes 0..15 / 16..31.
    float* ob = out + (size_t)(b * CCH) * HW + h * WW + w0;
#pragma unroll
    for (int v = 0; v < 8; ++v) {
        int m = v + 8 * lh;
        int oc0 = 0 * 16 + m;
        int oc1 = 1 * 16 + m;
        int oc2 = 2 * 16 + m;
        int oc3 = 3 * 16 + m;
        ob[oc0 * HW + lm] = acc0[v] + bias[oc0];
        ob[oc1 * HW + lm] = acc1[v] + bias[oc1];
        ob[oc2 * HW + lm] = acc2[v] + bias[oc2];
        ob[oc3 * HW + lm] = acc3[v] + bias[oc3];
    }
}

extern "C" void kernel_launch(void* const* d_in, const int* in_sizes, int n_in,
                              void* d_out, int out_size, void* d_ws, size_t ws_size,
                              hipStream_t stream) {
    const float* x    = (const float*)d_in[0];
    const float* sims = (const float*)d_in[1];
    const float* W    = (const float*)d_in[2];
    const float* bias = (const float*)d_in[3];
    float* out = (float*)d_out;

    float*  r_ws = (float*)d_ws;                 // R_ELEMS f32
    __bf16* wt2  = (__bf16*)(r_ws + R_ELEMS);    // WT2_ELEMS bf16 (~4.4 MB total)

    pack_w_kernel<<<(K2 * 4096 + 255) / 256, 256, 0, stream>>>(W, wt2);
    compute_r_kernel<<<(BSZ * HW + 255) / 256, 256, 0, stream>>>(sims, r_ws);
    sconv_wmma_kernel<<<NTILES / 8, 256, 0, stream>>>(x, r_ws, wt2, bias, out);
}
